// VariationalEncoder_71021579206869
// MI455X (gfx1250) — compile-verified, matrix operand order checked by reading, not run on hardware
//
#include <hip/hip_runtime.h>
#include <hip/hip_bf16.h>

#define TIN  256
#define THID 128
#define TOUT 64

typedef __attribute__((ext_vector_type(16))) __bf16 v16bf;
typedef __attribute__((ext_vector_type(8)))  float  v8f;

union AccU  { v8f   v; float        f[8]; };
union FragU { v16bf v; unsigned int u[8]; };

__device__ __forceinline__ unsigned int f2bf(float f) {
  // round-to-nearest-even f32 -> bf16 (low 16 bits of result)
  unsigned int u = __float_as_uint(f);
  u += 0x7FFFu + ((u >> 16) & 1u);
  return u >> 16;
}
__device__ __forceinline__ unsigned int pack2bf(float lo, float hi) {
  return f2bf(lo) | (f2bf(hi) << 16);
}

// ---------------- degree / norm ----------------
__global__ void deg_init_kernel(unsigned int* deg, int n) {
  int i = blockIdx.x * blockDim.x + threadIdx.x;
  if (i < n) deg[i] = 1u;  // self loop
}

__global__ void deg_accum_kernel(unsigned int* deg, const int* __restrict__ dst, int e) {
  int i = blockIdx.x * blockDim.x + threadIdx.x;
  if (i < e) atomicAdd(&deg[dst[i]], 1u);
}

__global__ void dinv_kernel(const unsigned int* __restrict__ deg, float* dinv, int n) {
  int i = blockIdx.x * blockDim.x + threadIdx.x;
  if (i < n) {
    float d = (float)deg[i];
    dinv[i] = (d > 0.f) ? rsqrtf(d) : 0.f;
  }
}

// ---------------- WMMA GEMM + self-loop/bias epilogue ----------------
// C[M,NC] = A[M,K] @ B[K,NC]  (A,B f32 in memory, bf16 in matrix cores, f32 acc)
// aggInit[m,n] = C[m,n]*dinv[m]^2 + bias[n]   (self-loop message + bias)
// Block tile 64x32, 8 waves, each wave one 16x16 WMMA tile, K-step 32.
// Epilogue transposes accumulators through LDS for coalesced b128 stores.
template <int K, int NC>
__global__ __launch_bounds__(256)
void gemm_bf16_wmma(const float* __restrict__ A, const float* __restrict__ B,
                    float* __restrict__ C, float* __restrict__ aggInit,
                    const float* __restrict__ dinv, const float* __restrict__ bias,
                    int M) {
  __shared__ unsigned short As[64 * 36];  // row-major [row][k], stride 36 halves
  __shared__ unsigned short Bs[32 * 36];  // col-major [col][k]
  __shared__ float          Cs[64 * 36];  // epilogue staging, stride 36 floats (16B aligned)

  const int tid  = threadIdx.x;
  const int row0 = blockIdx.x * 64;
  const int col0 = blockIdx.y * 32;

  const int wave = tid >> 5;
  const int lane = tid & 31;
  const int wm   = wave & 3;   // 4 sub-tiles along M
  const int wn   = wave >> 2;  // 2 sub-tiles along N
  const int hsel = lane >> 4;  // lane half (ISA A/B layout)
  const int l16  = lane & 15;

  const int aR  = tid >> 3;              // 0..31 (row, two passes cover 64)
  const int aKq = (tid & 7) * 4;         // k quad within 32
  const int bK  = tid >> 3;              // 0..31
  const int bNq = (tid & 7) * 4;         // n quad within 32

  AccU acc;
#pragma unroll
  for (int i = 0; i < 8; ++i) acc.f[i] = 0.f;

  for (int k0 = 0; k0 < K; k0 += 32) {
    // stage A tile 64x32: two float4 loads per thread, clamped row index
    // (garbage in OOB rows only feeds OOB output rows, which are masked at store)
#pragma unroll
    for (int j = 0; j < 2; ++j) {
      int r  = aR + 32 * j;
      int gr = row0 + r;
      int grc = (gr < M) ? gr : (M - 1);
      const float4 v = *(const float4*)&A[(size_t)grc * K + (k0 + aKq)];
      *(uint2*)&As[r * 36 + aKq] = make_uint2(pack2bf(v.x, v.y), pack2bf(v.z, v.w));
    }
    // stage B tile 32x32 (col-major in LDS): one float4 load per thread
    {
      const float4 v = *(const float4*)&B[(size_t)(k0 + bK) * NC + (col0 + bNq)];
      Bs[(bNq + 0) * 36 + bK] = (unsigned short)f2bf(v.x);
      Bs[(bNq + 1) * 36 + bK] = (unsigned short)f2bf(v.y);
      Bs[(bNq + 2) * 36 + bK] = (unsigned short)f2bf(v.z);
      Bs[(bNq + 3) * 36 + bK] = (unsigned short)f2bf(v.w);
    }
    __syncthreads();

    // pack fragments per ISA 16-bit 16x32 layout:
    // pair p<4: k = 2p + 8*half ; p>=4: k = 16 + 2(p-4) + 8*half
    FragU afrag, bfrag;
    const int arow = wm * 16 + l16;
    const int bcol = wn * 16 + l16;
#pragma unroll
    for (int p = 0; p < 8; ++p) {
      int k = ((p < 4) ? (2 * p) : (16 + 2 * (p - 4))) + 8 * hsel;
      afrag.u[p] = *(const unsigned int*)&As[arow * 36 + k];
      bfrag.u[p] = *(const unsigned int*)&Bs[bcol * 36 + k];
    }

    acc.v = __builtin_amdgcn_wmma_f32_16x16x32_bf16(
        false, afrag.v, false, bfrag.v, (short)0, acc.v, false, false);
    __syncthreads();
  }

  // ---- epilogue ----
  // 1) dump accumulators to LDS (C/D layout: lanes 0-15 M=r, lanes 16-31 M=r+8)
  {
    const int crow = wm * 16 + 8 * hsel;
    const int ccol = wn * 16 + l16;
#pragma unroll
    for (int r = 0; r < 8; ++r) Cs[(crow + r) * 36 + ccol] = acc.f[r];
  }
  __syncthreads();

  // 2) coalesced row-major writeback: thread -> (row, 4-col quad), b128 stores
  {
    const int cq = (tid & 7) * 4;
    const float4 bv = *(const float4*)&bias[col0 + cq];
#pragma unroll
    for (int j = 0; j < 2; ++j) {
      int r  = (tid >> 3) + 32 * j;
      int gr = row0 + r;
      if (gr < M) {
        float di = dinv[gr];
        float w  = di * di;
        const float4 v = *(const float4*)&Cs[r * 36 + cq];
        *(float4*)&C[(size_t)gr * NC + (col0 + cq)] = v;
        float4 o;
        o.x = fmaf(v.x, w, bv.x);
        o.y = fmaf(v.y, w, bv.y);
        o.z = fmaf(v.z, w, bv.z);
        o.w = fmaf(v.w, w, bv.w);
        *(float4*)&aggInit[(size_t)gr * NC + (col0 + cq)] = o;
      }
    }
  }
}

// ---------------- edge scatter (segment_sum via f32 atomics) ----------------
// one thread per (edge, 4-float chunk); chunks = F/4 = 1<<lgChunks
__global__ __launch_bounds__(256)
void edge_scatter_kernel(const int* __restrict__ src, const int* __restrict__ dst,
                         const float* __restrict__ dinv, const float* __restrict__ t,
                         float* __restrict__ agg, long long total, int F, int lgChunks) {
  long long gid = (long long)blockIdx.x * blockDim.x + threadIdx.x;
  if (gid >= total) return;
  int chunks = 1 << lgChunks;
  int e = (int)(gid >> lgChunks);
  int c = (int)(gid & (chunks - 1));
  int s = src[e], d = dst[e];
  float w = dinv[s] * dinv[d];
  const float4 m = *(const float4*)(t + (size_t)s * F + c * 4);
  float* o = agg + (size_t)d * F + c * 4;
  unsafeAtomicAdd(o + 0, m.x * w);
  unsafeAtomicAdd(o + 1, m.y * w);
  unsafeAtomicAdd(o + 2, m.z * w);
  unsafeAtomicAdd(o + 3, m.w * w);
}

__global__ void relu_kernel(const float* __restrict__ in, float* __restrict__ out, long long n) {
  long long i = (long long)blockIdx.x * blockDim.x + threadIdx.x;
  if (i < n) out[i] = fmaxf(in[i], 0.f);
}

// ---------------- launch ----------------
extern "C" void kernel_launch(void* const* d_in, const int* in_sizes, int n_in,
                              void* d_out, int out_size, void* d_ws, size_t ws_size,
                              hipStream_t stream) {
  (void)n_in; (void)out_size; (void)ws_size;
  const float* x   = (const float*)d_in[0];
  const float* W1  = (const float*)d_in[1];
  const float* b1  = (const float*)d_in[2];
  const float* Wmu = (const float*)d_in[3];
  const float* bmu = (const float*)d_in[4];
  const float* Wls = (const float*)d_in[5];
  const float* bls = (const float*)d_in[6];
  const int*   ei  = (const int*)d_in[7];

  const int N = in_sizes[0] / TIN;
  const int E = in_sizes[7] / 2;
  const int* srcp = ei;       // edge_index[0]
  const int* dstp = ei + E;   // edge_index[1]

  // workspace carve-out (256B aligned)
  size_t off = 0;
  auto take = [&](size_t bytes) -> void* {
    void* p = (char*)d_ws + off;
    off += (bytes + 255) & ~(size_t)255;
    return p;
  };
  unsigned int* deg = (unsigned int*)take((size_t)N * 4);
  float* dinv = (float*)take((size_t)N * 4);
  float* t1   = (float*)take((size_t)N * THID * 4);   // x@W1, later reused for h
  float* agg1 = (float*)take((size_t)N * THID * 4);
  float* tmu  = (float*)take((size_t)N * TOUT * 4);
  float* tls  = (float*)take((size_t)N * TOUT * 4);

  float* mu_out = (float*)d_out;
  float* ls_out = mu_out + (size_t)N * TOUT;

  const int tb = 256;
  // norm
  deg_init_kernel<<<(N + tb - 1) / tb, tb, 0, stream>>>(deg, N);
  deg_accum_kernel<<<(E + tb - 1) / tb, tb, 0, stream>>>(deg, dstp, E);
  dinv_kernel<<<(N + tb - 1) / tb, tb, 0, stream>>>(deg, dinv, N);

  // layer 1: t1 = x@W1 ; agg1 = self + b1 ; scatter edges ; h = relu(agg1) -> t1
  dim3 g1((N + 63) / 64, THID / 32);
  gemm_bf16_wmma<TIN, THID><<<g1, tb, 0, stream>>>(x, W1, t1, agg1, dinv, b1, N);

  long long th1 = (long long)E * (THID / 4);
  edge_scatter_kernel<<<(unsigned)((th1 + tb - 1) / tb), tb, 0, stream>>>(
      srcp, dstp, dinv, t1, agg1, th1, THID, 5);

  long long nh = (long long)N * THID;
  relu_kernel<<<(unsigned)((nh + tb - 1) / tb), tb, 0, stream>>>(agg1, t1, nh);

  // layer 2: mu / logstd straight into d_out (bias folded into init)
  dim3 g2((N + 63) / 64, TOUT / 32);
  gemm_bf16_wmma<THID, TOUT><<<g2, tb, 0, stream>>>(t1, Wmu, tmu, mu_out, dinv, bmu, N);
  gemm_bf16_wmma<THID, TOUT><<<g2, tb, 0, stream>>>(t1, Wls, tls, ls_out, dinv, bls, N);

  long long th2 = (long long)E * (TOUT / 4);
  edge_scatter_kernel<<<(unsigned)((th2 + tb - 1) / tb), tb, 0, stream>>>(
      srcp, dstp, dinv, tmu, mu_out, th2, TOUT, 4);
  edge_scatter_kernel<<<(unsigned)((th2 + tb - 1) / tb), tb, 0, stream>>>(
      srcp, dstp, dinv, tls, ls_out, th2, TOUT, 4);
}
